// ProbAttention_20890720927786
// MI455X (gfx1250) — matrix operand order repaired and checked by reading, not compiled, not run on hardware
//
#include <hip/hip_runtime.h>
#include <hip/hip_bf16.h>

// ---------------------------------------------------------------------------
// CDNA5 (gfx1250) RoPE multi-head attention, bf16 WMMA path, software-
// pipelined fragment loads. D_MODEL=1024, H=16, Dh=64, B=4, L=S=1024, M=4096.
// ---------------------------------------------------------------------------

typedef __attribute__((ext_vector_type(16))) __bf16        v16bf;
typedef __attribute__((ext_vector_type(8)))  float         v8f;
typedef __attribute__((ext_vector_type(4)))  unsigned int  u32x4;

union Frag16 { v16bf v; u32x4 q[2]; };

static __device__ __forceinline__ v8f wmma_bf16(v16bf a, v16bf b, v8f c) {
  // D = A(16x32 bf16) * B(32x16 bf16) + C(16x16 f32)
  return __builtin_amdgcn_wmma_f32_16x16x32_bf16(false, a, false, b,
                                                 (short)0, c, false, false);
}

// A fragment: lane holds row lo, 16 contiguous K at p[0..15]
static __device__ __forceinline__ void lda(Frag16& f, const __bf16* p) {
  f.q[0] = *(const u32x4*)p;
  f.q[1] = *(const u32x4*)(p + 8);
}
// B fragment (packed / contiguous-K): 16 contiguous bf16 per lane
static __device__ __forceinline__ void ldb(Frag16& f, const __bf16* p) {
  f.q[0] = *(const u32x4*)p;
  f.q[1] = *(const u32x4*)(p + 8);
}

// -------------------- pack fp32 activations -> bf16 row-major ---------------
__global__ void pack_f32_to_bf16(const float* __restrict__ x,
                                 __bf16* __restrict__ y, int n) {
  int i = blockIdx.x * blockDim.x + threadIdx.x;
  if (i < n) y[i] = (__bf16)x[i];
}

// ---- pack fp32 weight [K=1024][N=1024] into WMMA B-fragment layout ---------
// out[((kblk*64 + nblk)*32 + lane)*16 + e] = W[(kblk*32 + e + 16*hi)][nblk*16 + lane%16]
__global__ void pack_weight(const float* __restrict__ W, __bf16* __restrict__ out) {
  int idx = blockIdx.x * blockDim.x + threadIdx.x;     // < 2^20
  int e    =  idx        & 15;
  int lane = (idx >> 4)  & 31;
  int nblk = (idx >> 9)  & 63;
  int kblk =  idx >> 15;
  int k = kblk * 32 + e + ((lane & 16) ? 16 : 0);
  int n = nblk * 16 + (lane & 15);
  out[idx] = (__bf16)W[k * 1024 + n];
}

// -------------------- QKV projection + bias + RoPE --------------------------
// One wave computes a 32x64 tile (32 rows of B*L, one head's 64 columns),
// 8 WMMAs per K-step, double-buffered fragment loads.
__global__ __launch_bounds__(32) void proj_rope_kernel(
    const __bf16* __restrict__ X,      // [4096][1024] bf16
    const __bf16* __restrict__ Wp,     // packed weight fragments
    const float*  __restrict__ bias,   // [1024]
    __bf16* __restrict__ out,
    int applyRope, int storeVT) {
  const int lane = threadIdx.x;
  const int lo   = lane & 15;
  const int hi8  = (lane & 16) ? 8 : 0;
  const int gm0  = blockIdx.x * 32;        // global row tile (B*L)
  const int h    = blockIdx.y;             // head
  const int b    = gm0 >> 10;
  const int l0   = gm0 & 1023;

  const __bf16* ap0 = X + (size_t)(gm0 + lo) * 1024 + ((lane & 16) ? 16 : 0);
  const __bf16* wp0 = Wp + (size_t)((h * 4) * 32 + lane) * 16;
  const size_t  WSTEP = (size_t)64 * 32 * 16;   // packed elems per k-block

  Frag16 abuf[2][2], bbuf[2][4];
  lda(abuf[0][0], ap0);
  lda(abuf[0][1], ap0 + 16 * 1024);
#pragma unroll
  for (int f = 0; f < 4; ++f) ldb(bbuf[0][f], wp0 + f * 512);

  v8f acc[2][4] = {};
#pragma unroll 2
  for (int kb = 0; kb < 32; ++kb) {
    const int cur = kb & 1, nxt = cur ^ 1;
    if (kb + 1 < 32) {                        // prefetch next stage
      const __bf16* a = ap0 + (kb + 1) * 32;
      lda(abuf[nxt][0], a);
      lda(abuf[nxt][1], a + 16 * 1024);
      const __bf16* w = wp0 + (size_t)(kb + 1) * WSTEP;
#pragma unroll
      for (int f = 0; f < 4; ++f) ldb(bbuf[nxt][f], w + f * 512);
    }
#pragma unroll
    for (int f = 0; f < 4; ++f) {
#pragma unroll
      for (int rh = 0; rh < 2; ++rh)
        acc[rh][f] = wmma_bf16(abuf[cur][rh].v, bbuf[cur][f].v, acc[rh][f]);
    }
  }

  const int bh = b * 16 + h;
#pragma unroll
  for (int rh = 0; rh < 2; ++rh) {
    // bias
#pragma unroll
    for (int f = 0; f < 4; ++f) {
      float bv = bias[h * 64 + f * 16 + lo];
#pragma unroll
      for (int r = 0; r < 8; ++r) acc[rh][f][r] += bv;
    }
    if (applyRope) {
      // pairs (f, f+2): d and d+32 share the same lane/row slot.
#pragma unroll
      for (int f = 0; f < 2; ++f) {
        float inv = __expf(-0.28782314f * (float)(f * 16 + lo)); // exp(-ln(1e4)/32*d)
#pragma unroll
        for (int r = 0; r < 8; ++r) {
          float pos = (float)(l0 + rh * 16 + r + hi8);
          float s, c;
          __sincosf(pos * inv, &s, &c);
          float t1 = acc[rh][f][r], t2 = acc[rh][f + 2][r];
          acc[rh][f][r]     = t1 * c - t2 * s;
          acc[rh][f + 2][r] = t1 * s + t2 * c;
        }
      }
    }
#pragma unroll
    for (int f = 0; f < 4; ++f) {
      int d = f * 16 + lo;
#pragma unroll
      for (int r = 0; r < 8; ++r) {
        int l = l0 + rh * 16 + r + hi8;
        __bf16 val = (__bf16)acc[rh][f][r];
        if (storeVT) out[(size_t)(bh * 64 + d) * 1024 + l] = val;  // V^T [bh][d][s]
        else         out[(size_t)(bh * 1024 + l) * 64 + d] = val;  // Q/K [bh][l][d]
      }
    }
  }
}

// -------------------- flash attention (one wave = 16 q-rows of one head) ----
__global__ __launch_bounds__(32) void attn_kernel(
    const __bf16* __restrict__ qh,   // [64][1024][64]
    const __bf16* __restrict__ kh,   // [64][1024][64]
    const __bf16* __restrict__ vT,   // [64][64][1024]
    __bf16* __restrict__ ctx) {      // [4096][1024]
  __shared__ float  sS[16][32];
  __shared__ __bf16 sP[16][32];
  __shared__ float  sAlpha[16];
  __shared__ float  sL[16];

  const int lane = threadIdx.x;
  const int lo   = lane & 15;
  const int hi8  = (lane & 16) ? 8 : 0;
  const int hi16 = (lane & 16) ? 16 : 0;
  const int tile = blockIdx.x;            // 0..4095
  const int q0   = (tile & 63) * 16;
  const int bh   = tile >> 6;
  const int b    = bh >> 4, h = bh & 15;

  // Q A-fragments for Dh halves (K=0..31 and K=32..63), reused all iterations.
  Frag16 qa[2];
  {
    const __bf16* qrow = qh + (size_t)(bh * 1024 + q0 + lo) * 64;
    lda(qa[0], qrow + hi16);
    lda(qa[1], qrow + 32 + hi16);
  }

  // K score-B fragments: [0]=cols t0 d0..31, [1]=cols t0 d32..63, [2],[3]=cols t1
  auto ldk = [&](Frag16 kb[4], int s0) {
#pragma unroll
    for (int t = 0; t < 2; ++t) {
      const __bf16* kp = kh + (size_t)(bh * 1024 + s0 + t * 16 + lo) * 64 + hi16;
      ldb(kb[t * 2 + 0], kp);
      ldb(kb[t * 2 + 1], kp + 32);
    }
  };

  Frag16 kbuf[2][4];
  ldk(kbuf[0], 0);

  v8f acc[4] = {};
  float m_i = -1e30f, l_i = 0.0f;       // per-row stats, owned by lanes 0..15

#pragma unroll 2
  for (int it = 0; it < 32; ++it) {
    const int s0 = it * 32;
    const int cur = it & 1, nxt = cur ^ 1;

    // ---- V fragments for this chunk (independent of softmax; hide latency) -
    Frag16 vb[4];
#pragma unroll
    for (int f = 0; f < 4; ++f)
      ldb(vb[f], vT + (size_t)(bh * 64 + f * 16 + lo) * 1024 + s0 + hi16);

    // ---- scores: two 16x16 tiles over 32 key columns -----------------------
    v8f c0 = {}, c1 = {};
    c0 = wmma_bf16(qa[0].v, kbuf[cur][0].v, c0);
    c0 = wmma_bf16(qa[1].v, kbuf[cur][1].v, c0);
    c1 = wmma_bf16(qa[0].v, kbuf[cur][2].v, c1);
    c1 = wmma_bf16(qa[1].v, kbuf[cur][3].v, c1);

    // ---- prefetch next chunk's K fragments (overlaps softmax) --------------
    if (it + 1 < 32) ldk(kbuf[nxt], s0 + 32);

    // ---- scores -> LDS (transpose point for softmax + A-fragment repack) ---
#pragma unroll
    for (int r = 0; r < 8; ++r) {
      int m = r + hi8;
      sS[m][lo]      = c0[r];
      sS[m][lo + 16] = c1[r];
    }
    // ---- online softmax: lane r (<16) owns row r ---------------------------
    if (lane < 16) {
      float mx = m_i;
      for (int j = 0; j < 32; ++j) mx = fmaxf(mx, sS[lane][j] * 0.125f);
      float alpha = __expf(m_i - mx);
      float sum = 0.0f;
      for (int j = 0; j < 32; ++j) {
        float p = __expf(sS[lane][j] * 0.125f - mx);
        sum += p;
        sP[lane][j] = (__bf16)p;
      }
      l_i = l_i * alpha + sum;
      m_i = mx;
      sAlpha[lane] = alpha;
    }
    // ---- rescale output accumulators by alpha ------------------------------
#pragma unroll
    for (int r = 0; r < 8; ++r) {
      float a = sAlpha[r + hi8];
#pragma unroll
      for (int f = 0; f < 4; ++f) acc[f][r] *= a;
    }
    // ---- P as 16x32 A-fragment straight out of LDS -------------------------
    Frag16 pa;
    lda(pa, &sP[lo][hi16]);
    // ---- acc += P @ V ------------------------------------------------------
#pragma unroll
    for (int f = 0; f < 4; ++f) acc[f] = wmma_bf16(pa.v, vb[f].v, acc[f]);
  }

  if (lane < 16) sL[lane] = l_i;
#pragma unroll
  for (int r = 0; r < 8; ++r) {
    int m = r + hi8;
    float inv = 1.0f / sL[m];
    int row = q0 + m;
#pragma unroll
    for (int f = 0; f < 4; ++f) {
      ctx[(size_t)(b * 1024 + row) * 1024 + h * 64 + f * 16 + lo] =
          (__bf16)(acc[f][r] * inv);
    }
  }
}

// -------------------- output projection (fp32 out) --------------------------
// 32x64 tile per wave, double-buffered like proj.
__global__ __launch_bounds__(32) void out_proj_kernel(
    const __bf16* __restrict__ Xc, const __bf16* __restrict__ Wp,
    const float* __restrict__ bias, float* __restrict__ out) {
  const int lane = threadIdx.x;
  const int lo   = lane & 15;
  const int hi8  = (lane & 16) ? 8 : 0;
  const int gm0  = blockIdx.x * 32;
  const int nt   = blockIdx.y;             // 64-column tile

  const __bf16* ap0 = Xc + (size_t)(gm0 + lo) * 1024 + ((lane & 16) ? 16 : 0);
  const __bf16* wp0 = Wp + (size_t)((nt * 4) * 32 + lane) * 16;
  const size_t  WSTEP = (size_t)64 * 32 * 16;

  Frag16 abuf[2][2], bbuf[2][4];
  lda(abuf[0][0], ap0);
  lda(abuf[0][1], ap0 + 16 * 1024);
#pragma unroll
  for (int f = 0; f < 4; ++f) ldb(bbuf[0][f], wp0 + f * 512);

  v8f acc[2][4] = {};
#pragma unroll 2
  for (int kb = 0; kb < 32; ++kb) {
    const int cur = kb & 1, nxt = cur ^ 1;
    if (kb + 1 < 32) {
      const __bf16* a = ap0 + (kb + 1) * 32;
      lda(abuf[nxt][0], a);
      lda(abuf[nxt][1], a + 16 * 1024);
      const __bf16* w = wp0 + (size_t)(kb + 1) * WSTEP;
#pragma unroll
      for (int f = 0; f < 4; ++f) ldb(bbuf[nxt][f], w + f * 512);
    }
#pragma unroll
    for (int f = 0; f < 4; ++f) {
#pragma unroll
      for (int rh = 0; rh < 2; ++rh)
        acc[rh][f] = wmma_bf16(abuf[cur][rh].v, bbuf[cur][f].v, acc[rh][f]);
    }
  }

#pragma unroll
  for (int rh = 0; rh < 2; ++rh) {
#pragma unroll
    for (int f = 0; f < 4; ++f) {
      int n = nt * 64 + f * 16 + lo;
      float bv = bias[n];
#pragma unroll
      for (int r = 0; r < 8; ++r) {
        int m = gm0 + rh * 16 + r + hi8;
        out[(size_t)m * 1024 + n] = acc[rh][f][r] + bv;
      }
    }
  }
}

// ---------------------------------------------------------------------------
extern "C" void kernel_launch(void* const* d_in, const int* in_sizes, int n_in,
                              void* d_out, int out_size, void* d_ws, size_t ws_size,
                              hipStream_t stream) {
  (void)in_sizes; (void)n_in; (void)out_size; (void)ws_size;
  const float* queries = (const float*)d_in[0];
  const float* keys    = (const float*)d_in[1];
  const float* values  = (const float*)d_in[2];
  const float* Wq = (const float*)d_in[3]; const float* bq = (const float*)d_in[4];
  const float* Wk = (const float*)d_in[5]; const float* bk = (const float*)d_in[6];
  const float* Wv = (const float*)d_in[7]; const float* bv = (const float*)d_in[8];
  const float* Wo = (const float*)d_in[9]; const float* bo = (const float*)d_in[10];
  float* out = (float*)d_out;

  char* ws = (char*)d_ws;
  const size_t MB = 1024ull * 1024ull;
  __bf16* xq  = (__bf16*)(ws +  0 * MB);   // 8 MB each activation
  __bf16* xk  = (__bf16*)(ws +  8 * MB);
  __bf16* xv  = (__bf16*)(ws + 16 * MB);
  __bf16* wqp = (__bf16*)(ws + 24 * MB);   // 2 MB each packed weight
  __bf16* wkp = (__bf16*)(ws + 26 * MB);
  __bf16* wvp = (__bf16*)(ws + 28 * MB);
  __bf16* wop = (__bf16*)(ws + 30 * MB);
  __bf16* qhh = (__bf16*)(ws + 32 * MB);   // [64][1024][64]
  __bf16* khh = (__bf16*)(ws + 40 * MB);
  __bf16* vt  = (__bf16*)(ws + 48 * MB);   // [64][64][1024]
  __bf16* ctx = (__bf16*)(ws + 56 * MB);   // [4096][1024]

  const int nx = 4 * 1024 * 1024;          // B*L*D
  pack_f32_to_bf16<<<nx / 256, 256, 0, stream>>>(queries, xq, nx);
  pack_f32_to_bf16<<<nx / 256, 256, 0, stream>>>(keys,    xk, nx);
  pack_f32_to_bf16<<<nx / 256, 256, 0, stream>>>(values,  xv, nx);
  const int nw = 1024 * 1024;
  pack_weight<<<nw / 256, 256, 0, stream>>>(Wq, wqp);
  pack_weight<<<nw / 256, 256, 0, stream>>>(Wk, wkp);
  pack_weight<<<nw / 256, 256, 0, stream>>>(Wv, wvp);
  pack_weight<<<nw / 256, 256, 0, stream>>>(Wo, wop);

  dim3 pg(128, 16);
  proj_rope_kernel<<<pg, 32, 0, stream>>>(xq, wqp, bq, qhh, 1, 0);
  proj_rope_kernel<<<pg, 32, 0, stream>>>(xk, wkp, bk, khh, 1, 0);
  proj_rope_kernel<<<pg, 32, 0, stream>>>(xv, wvp, bv, vt,  0, 1);

  attn_kernel<<<4096, 32, 0, stream>>>(qhh, khh, vt, ctx);

  out_proj_kernel<<<pg, 32, 0, stream>>>(ctx, wop, bo, out);
}
